// MSDAv4_46394236731952
// MI455X (gfx1250) — matrix operand order repaired and checked by reading, not compiled, not run on hardware
//
#include <hip/hip_runtime.h>

// MSDA (multi-scale deformable attention) for MI455X / gfx1250.
// GEMMs via v_wmma_f32_16x16x32_f16 (f32 accumulate), A staged in LDS with
// async-to-LDS loads (ASYNCcnt) when available, gathers lane-coalesced.

typedef __attribute__((ext_vector_type(16))) _Float16 v16h;
typedef __attribute__((ext_vector_type(8)))  float    v8f;
typedef int v4i_vs __attribute__((vector_size(16)));   // matches builtin param

#define BATCH  2
#define NPTS   5456
#define BN_    (BATCH * NPTS)   // 10912 = 682 * 16
#define HNUM   8
#define HD     32
#define LNUM   5
#define PNUM   4
#define KDIM   256
#define NT     4                // N-tiles (16 cols each) per wave

#if defined(__has_builtin)
#if __has_builtin(__builtin_amdgcn_global_load_async_to_lds_b128)
#define HAVE_ASYNC_LDS 1
#endif
#endif

typedef __attribute__((address_space(1))) v4i_vs* gv4i_p;
typedef __attribute__((address_space(3))) v4i_vs* lv4i_p;

__device__ __forceinline__ unsigned int pack2h(float a, float b) {
  union { _Float16 h[2]; unsigned int u; } p;
  p.h[0] = (_Float16)a; p.h[1] = (_Float16)b;
  return p.u;
}

// C[m,n] = sum_k A[m,k] * W[n,k] + bias[n]
// A: [M,256] f32, W: [Nout,256] f32 row-major, C: [M,Nout] f32.
// Block = 128 threads (4 waves) covering a 16(M) x 256(N) output tile:
// wave w owns 4 consecutive 16x16 tiles (4 accumulators, shared A fragment).
// The full 16x256 A row-block is staged once in LDS, so the k-loop has no
// barriers and fully unrolls into 32 back-to-back WMMAs. Out-of-range tiles
// are CLAMPED to the last valid tile (redundant identical math, loads always
// in-bounds) so the hot loop is branch-free; only the epilogue store is
// guarded by tile ownership.
__global__ __launch_bounds__(128)
void gemm_f16_wmma(const float* __restrict__ A, const float* __restrict__ W,
                   const float* __restrict__ bias, float* __restrict__ C,
                   int M, int Nout) {
  __shared__ float ldsA[16 * KDIM];   // 16 KB (of 320 KB/WGP)
  const int tid    = threadIdx.x;
  const int lane   = tid & 31;
  const int wave   = tid >> 5;
  const int m0     = blockIdx.x * 16;
  const int ntiles = Nout >> 4;
  const int hi     = lane >> 4;       // half-wave select
  const int idx    = lane & 15;       // row (A) / col (B,C) within tile

  // ---- Stage the whole 16 x 256 f32 A row-block into LDS ----
  {
    const float* src = &A[(size_t)m0 * KDIM + tid * 32];
    float*       dst = &ldsA[tid * 32];
#ifdef HAVE_ASYNC_LDS
#pragma unroll
    for (int i = 0; i < 8; ++i) {
      __builtin_amdgcn_global_load_async_to_lds_b128(
          (gv4i_p)(src + i * 4), (lv4i_p)(dst + i * 4), 0, 0);
    }
#if __has_builtin(__builtin_amdgcn_s_wait_asynccnt)
    __builtin_amdgcn_s_wait_asynccnt(0);
#else
    asm volatile("s_wait_asynccnt 0x0" ::: "memory");
#endif
#else
#pragma unroll
    for (int i = 0; i < 8; ++i) {
      *reinterpret_cast<float4*>(dst + i * 4) =
          *reinterpret_cast<const float4*>(src + i * 4);
    }
#endif
  }
  __syncthreads();

  // Tile assignment (all wave-uniform): clamp OOB tiles onto the last tile.
  int  ncol[NT];
  bool owner[NT];
#pragma unroll
  for (int t = 0; t < NT; ++t) {
    const int tt = blockIdx.y * 16 + wave * NT + t;
    owner[t] = (tt < ntiles);
    ncol[t]  = (owner[t] ? tt : (ntiles - 1)) * 16;
  }

  v8f acc[NT] = {};
#pragma unroll
  for (int k0 = 0; k0 < KDIM; k0 += 32) {
    // A 16x32 f16 fragment (ISA 7.12.2): lane<16 -> K {0..7,16..23},
    // lane>=16 -> K {8..15,24..31}; 2 consecutive K per VGPR.
    union { unsigned int u[8]; v16h v; } af;
#pragma unroll
    for (int v = 0; v < 8; ++v) {
      const int ka = 2 * v + hi * 8 + ((v < 4) ? 0 : 8);
      const float2 ap =
          *reinterpret_cast<const float2*>(&ldsA[idx * KDIM + k0 + ka]);
      af.u[v] = pack2h(ap.x, ap.y);
    }
#pragma unroll
    for (int t = 0; t < NT; ++t) {
      // B 32x16: lane<16 -> K {0..7,16..23} of column idx, lane>=16 -> K {8..15,24..31}.
      union { unsigned int u[8]; v16h v; } bf;
#pragma unroll
      for (int v = 0; v < 8; ++v) {
        const int kb = 2 * v + hi * 8 + ((v < 4) ? 0 : 8);
        const float2 wv = *reinterpret_cast<const float2*>(
            &W[(size_t)(ncol[t] + idx) * KDIM + k0 + kb]);
        bf.u[v] = pack2h(wv.x, wv.y);
      }
      acc[t] = __builtin_amdgcn_wmma_f32_16x16x32_f16(
          false, af.v, false, bf.v, (short)0, acc[t], false, false);
    }
  }

#pragma unroll
  for (int t = 0; t < NT; ++t) {
    if (owner[t]) {   // wave-uniform store guard only in the epilogue
      const float bv = bias[ncol[t] + idx];
#pragma unroll
      for (int i = 0; i < 8; ++i) {
        // C/D layout: VGPR i -> row m0 + i + 8*hi, col ncol + idx.
        C[(size_t)(m0 + i + hi * 8) * Nout + ncol[t] + idx] = acc[t][i] + bv;
      }
    }
  }
}

// One wave per (b, n, h); lane = head-dim channel d (HD == warp == 32).
// Scalar sampling math is wave-uniform; each gather corner is a coalesced
// 128B load of val[(b*N + sidx)*256 + h*32 + d], served from L2 (val = 11MB).
__global__ __launch_bounds__(256)
void msda_sample(const float* __restrict__ offs,   // [BN,480] = (H,L,P,3)
                 const float* __restrict__ logits, // [BN,160] = (H,L*P)
                 const float* __restrict__ priors, // [B,N,L,2]
                 const float* __restrict__ val,    // [BN,256] = (.., H,HD)
                 float* __restrict__ weighted) {   // [BN,256]
  const int lane = threadIdx.x & 31;
  const int wid  = blockIdx.x * 8 + (threadIdx.x >> 5);
  const int b    = wid / (NPTS * HNUM);
  const int rem  = wid - b * NPTS * HNUM;
  const int n    = rem / HNUM;
  const int h    = rem - n * HNUM;
  const size_t row = (size_t)b * NPTS + n;

  // Softmax over the 20 logits (fully unrolled -> registers, no scratch).
  const float* lg = logits + row * 160 + h * 20;
  float w[LNUM * PNUM];
  float mx = -1e30f;
#pragma unroll
  for (int j = 0; j < LNUM * PNUM; ++j) { w[j] = lg[j]; mx = fmaxf(mx, w[j]); }
  float s = 0.f;
#pragma unroll
  for (int j = 0; j < LNUM * PNUM; ++j) { w[j] = __expf(w[j] - mx); s += w[j]; }
  const float inv = 1.f / s;

  const float* of = offs + row * 480 + h * 60;
  const float* vb = val + (size_t)b * NPTS * 256 + h * HD + lane;
  float acc = 0.f;

#pragma unroll
  for (int l = 0; l < LNUM; ++l) {
    const float px = priors[(row * LNUM + l) * 2 + 0];
    const float py = priors[(row * LNUM + l) * 2 + 1];
    const float sc = 1.0f / (float)(64 >> l);     // map sizes 64,32,16,8,4
#pragma unroll
    for (int p = 0; p < PNUM; ++p) {
      const float x = px + of[l * 12 + p * 3 + 0] * sc;
      const float y = py + of[l * 12 + p * 3 + 1] * sc;
      const float z = (float)l * 0.25f + of[l * 12 + p * 3 + 2];
      const float zc = fminf(fmaxf(z, 0.f), 1.f) * (float)(LNUM - 1);
      int z0 = (int)floorf(zc);
      if (z0 > LNUM - 2) z0 = LNUM - 2;
      if (z0 < 0) z0 = 0;
      const float wz = zc - (float)z0;
      const float aw = w[l * PNUM + p] * inv;
#pragma unroll
      for (int t = 0; t < 2; ++t) {
        const int   lev = z0 + t;
        const float wl  = aw * (t == 0 ? (1.f - wz) : wz);
        const int Wd   = 64 >> lev;                            // 64,32,16,8,4
        const int off0 = (16384 - (16384 >> (2 * lev))) / 3;   // 0,4096,5120,..
        const float xf = x * (float)Wd - 0.5f;
        const float yf = y * (float)Wd - 0.5f;
        const float x0f = floorf(xf), y0f = floorf(yf);
        const int x0 = (int)x0f, y0 = (int)y0f;
        const float dx = xf - x0f, dy = yf - y0f;
#pragma unroll
        for (int c = 0; c < 4; ++c) {
          const int ix = x0 + (c & 1);
          const int iy = y0 + (c >> 1);
          const float cw = ((c & 1) ? dx : 1.f - dx) *
                           ((c >> 1) ? dy : 1.f - dy);
          if (ix >= 0 && ix < Wd && iy >= 0 && iy < Wd) {
            const int sidx = off0 + iy * Wd + ix;
            acc += wl * cw * vb[(size_t)sidx * 256];
          }
        }
      }
    }
  }
  weighted[row * 256 + h * HD + lane] = acc;
}

extern "C" void kernel_launch(void* const* d_in, const int* in_sizes, int n_in,
                              void* d_out, int out_size, void* d_ws, size_t ws_size,
                              hipStream_t stream) {
  (void)in_sizes; (void)n_in; (void)out_size; (void)ws_size;
  const float* in_feats = (const float*)d_in[0];
  const float* priors   = (const float*)d_in[1];
  const float* sfeats   = (const float*)d_in[2];
  const float* W_off    = (const float*)d_in[3];
  const float* b_off    = (const float*)d_in[4];
  const float* W_attn   = (const float*)d_in[5];
  const float* b_attn   = (const float*)d_in[6];
  const float* W_val    = (const float*)d_in[7];
  const float* b_val    = (const float*)d_in[8];
  const float* W_out    = (const float*)d_in[9];
  const float* b_out    = (const float*)d_in[10];
  float* out = (float*)d_out;

  float* ws     = (float*)d_ws;
  float* offs   = ws;                               // BN*480
  float* logit  = offs  + (size_t)BN_ * 480;        // BN*160
  float* valb   = logit + (size_t)BN_ * 160;        // BN*256
  float* wtd    = valb  + (size_t)BN_ * 256;        // BN*256

  const int mt = BN_ / 16;  // 682
  // grid.y = ceil((Nout/16) / 16): each block covers 256 output columns.
  gemm_f16_wmma<<<dim3(mt, 2), 128, 0, stream>>>(in_feats, W_off,  b_off,  offs,  BN_, 480);
  gemm_f16_wmma<<<dim3(mt, 1), 128, 0, stream>>>(in_feats, W_attn, b_attn, logit, BN_, 160);
  gemm_f16_wmma<<<dim3(mt, 1), 128, 0, stream>>>(sfeats,   W_val,  b_val,  valb,  BN_, 256);
  msda_sample<<<dim3(BN_ * HNUM / 8), 256, 0, stream>>>(offs, logit, priors, valb, wtd);
  gemm_f16_wmma<<<dim3(mt, 1), 128, 0, stream>>>(wtd,      W_out,  b_out,  out,   BN_, 256);
}
